// VectorQuantizerEMA_11020886082297
// MI455X (gfx1250) — matrix operand order repaired and checked by reading, not compiled, not run on hardware
//
#include <hip/hip_runtime.h>
#include <hip/hip_bf16.h>

typedef __attribute__((ext_vector_type(16))) __bf16 v16bf;
typedef __attribute__((ext_vector_type(8)))  float  v8f;

#define KCODES 2048
#define DDIM   256
#define BATCH  16
#define HWSZ   4096              // 64*64
#define NVEC   65536             // BATCH*HWSZ
#define DECAYF 0.99f
#define EPSF   1e-05f
#define ROWS   32                // query rows per block (2 WMMA row-tiles)
#define LDS_STRIDE 264           // 256 + 8 halves: 528B rows, 16B aligned, bank-staggered

__device__ __forceinline__ unsigned short f2bf(float f) {
    unsigned u = __float_as_uint(f);
    u += 0x7FFFu + ((u >> 16) & 1u);     // round-to-nearest-even
    return (unsigned short)(u >> 16);
}

// ---------------- kernel 1: zero stats scratch ----------------
__global__ void k_zero(float* p, int n) {
    int i = blockIdx.x * 256 + threadIdx.x;
    if (i < n) p[i] = 0.0f;
}

// ---------------- kernel 2: bf16 codebook + ||c||^2 ----------------
__global__ __launch_bounds__(256) void k_pre(const float* __restrict__ cb,
                                             unsigned short* __restrict__ cbb,
                                             float* __restrict__ cnorm) {
    int k = blockIdx.x, d = threadIdx.x;
    float v = cb[k * DDIM + d];
    cbb[k * DDIM + d] = f2bf(v);
    __shared__ float red[256];
    red[d] = v * v;
    __syncthreads();
    for (int s = 128; s > 0; s >>= 1) {
        if (d < s) red[d] += red[d + s];
        __syncthreads();
    }
    if (d == 0) cnorm[k] = red[0];
}

// ---------------- kernel 3: WMMA distance + argmin ----------------
// Block = 4 waves, 32 query rows. Each wave keeps BOTH 16x256 A-fragment sets
// in VGPRs and scans 32 of the 128 16-code column tiles; B is streamed from
// the bf16 codebook (L2-resident) through a depth-2 register pipeline with
// global_prefetch lookahead on the next tile's slab.
__global__ __launch_bounds__(128) void k_argmin(const float* __restrict__ ze,
                                                const unsigned short* __restrict__ cbb,
                                                const float* __restrict__ cnorm,
                                                int* __restrict__ wsIdx,
                                                float* __restrict__ outIdx,
                                                float* __restrict__ counts) {
    __shared__ unsigned short ldsA[ROWS * LDS_STRIDE];
    __shared__ float redv[4][ROWS];
    __shared__ int   redi[4][ROWS];

    const int tid  = threadIdx.x;
    const int lane = tid & 31;
    const int wave = tid >> 5;
    const int n0   = blockIdx.x * ROWS;     // first of 32 query rows
    const int b    = n0 >> 12;              // n0 / 4096
    const int hwb  = n0 & (HWSZ - 1);

    // stage 32x256 fp32 tile -> LDS bf16 (row r = query, col d); 128B segments
    for (int it = 0; it < 64; ++it) {
        int lin = it * 128 + tid;           // 0..8191
        int d = lin >> 5;
        int r = lin & 31;
        float v = ze[(b * DDIM + d) * HWSZ + hwb + r];
        ldsA[r * LDS_STRIDE + d] = f2bf(v);
    }
    __syncthreads();

    const int m = lane & 15;                // A: row M / B: col N within tile
    const int h = lane >> 4;                // K-half selector

    // A fragments, resident in VGPRs (two row-tile sets).
    // 16-bit A 16x32 layout (matches B): lane(m,h) holds K = [32kk + 16h .. +15] (contiguous)
    v16bf af0[8], af1[8];
    {
        const unsigned short* r0 = &ldsA[m * LDS_STRIDE];
        const unsigned short* r1 = &ldsA[(16 + m) * LDS_STRIDE];
#pragma unroll
        for (int kk = 0; kk < 8; ++kk) {
            v16bf a;
            __builtin_memcpy(&a, r0 + kk * 32 + h * 16, 32);
            af0[kk] = a;
            v16bf a1;
            __builtin_memcpy(&a1, r1 + kk * 32 + h * 16, 32);
            af1[kk] = a1;
        }
    }

    float minv[16];
    int   mini[16];
#pragma unroll
    for (int i = 0; i < 16; ++i) { minv[i] = 3.4e38f; mini[i] = 0; }

    // each wave scans 32 of the 128 column tiles (16 codes each)
    for (int ct = wave; ct < KCODES / 16; ct += 4) {
        // prefetch the 8KB B slab of tile ct+4 (clamped: no divergent guard):
        // 32 lanes x 2 points at 128B stride cover every cacheline
        int pct = ct + 4 < KCODES / 16 ? ct + 4 : KCODES / 16 - 1;
        const char* pf = (const char*)(cbb + pct * 16 * DDIM) + lane * 256;
        __builtin_prefetch(pf, 0, 3);
        __builtin_prefetch(pf + 128, 0, 3);

        // B 32x16 bf16: lane(n=m,h) holds codebook[ct*16+n][32kk + 16h .. +15] (contiguous)
        const unsigned short* bp = cbb + (ct * 16 + m) * DDIM + h * 16;
        v8f c0 = {0.f, 0.f, 0.f, 0.f, 0.f, 0.f, 0.f, 0.f};
        v8f c1 = {0.f, 0.f, 0.f, 0.f, 0.f, 0.f, 0.f, 0.f};

        // depth-2 register pipeline on B so loads run ahead of the WMMA pipe
        v16bf b0, b1;
        __builtin_memcpy(&b0, bp, 32);
        __builtin_memcpy(&b1, bp + 32, 32);
#pragma unroll
        for (int kk = 0; kk < 8; ++kk) {
            v16bf bn = b1;
            if (kk < 6) __builtin_memcpy(&bn, bp + (kk + 2) * 32, 32);
            c0 = __builtin_amdgcn_wmma_f32_16x16x32_bf16(
                     false, af0[kk], false, b0, (short)0, c0, false, false);
            c1 = __builtin_amdgcn_wmma_f32_16x16x32_bf16(
                     false, af1[kk], false, b0, (short)0, c1, false, false);
            b0 = b1;
            b1 = bn;
        }
        float cn  = cnorm[ct * 16 + m];
        int  code = ct * 16 + m;
#pragma unroll
        for (int i = 0; i < 8; ++i) {
            float d0 = cn - 2.0f * c0[i];        // ||x||^2 constant per row: irrelevant to argmin
            float d1 = cn - 2.0f * c1[i];
            if (d0 < minv[i])     { minv[i]     = d0; mini[i]     = code; }
            if (d1 < minv[8 + i]) { minv[8 + i] = d1; mini[8 + i] = code; }
        }
    }

    // reduce across the 16 lanes of each half-wave (different codes, same rows)
#pragma unroll
    for (int i = 0; i < 16; ++i) {
        for (int mk = 1; mk < 16; mk <<= 1) {
            float ov = __shfl_xor(minv[i], mk, 32);
            int   oi = __shfl_xor(mini[i], mk, 32);
            if (ov < minv[i] || (ov == minv[i] && oi < mini[i])) { minv[i] = ov; mini[i] = oi; }
        }
    }
    // C layout: lane 0 holds M=i (vgpr i), lane 16 holds M=8+i
    if (lane == 0) {
#pragma unroll
        for (int i = 0; i < 8; ++i) {
            redv[wave][i]      = minv[i];      redi[wave][i]      = mini[i];       // rows 0..7
            redv[wave][16 + i] = minv[8 + i];  redi[wave][16 + i] = mini[8 + i];   // rows 16..23
        }
    }
    if (lane == 16) {
#pragma unroll
        for (int i = 0; i < 8; ++i) {
            redv[wave][8 + i]  = minv[i];      redi[wave][8 + i]  = mini[i];       // rows 8..15
            redv[wave][24 + i] = minv[8 + i];  redi[wave][24 + i] = mini[8 + i];   // rows 24..31
        }
    }
    __syncthreads();

    if (tid < ROWS) {
        float bv = redv[0][tid];
        int   bi = redi[0][tid];
#pragma unroll
        for (int w = 1; w < 4; ++w) {
            float ov = redv[w][tid];
            int   oi = redi[w][tid];
            if (ov < bv || (ov == bv && oi < bi)) { bv = ov; bi = oi; }
        }
        int n = n0 + tid;
        wsIdx[n]  = bi;
        outIdx[n] = (float)bi;
        atomicAdd(&counts[bi], 1.0f);
    }
}

// ---------------- kernel 4: z_q gather, dw scatter, commit loss ----------------
__global__ __launch_bounds__(256) void k_zq(const float* __restrict__ ze,
                                            const float* __restrict__ cb,
                                            const int* __restrict__ idx,
                                            float* __restrict__ zq,
                                            float* __restrict__ dw,
                                            float* __restrict__ lossAcc) {
    int plane = blockIdx.x;                 // = b*256 + d
    int d = plane & (DDIM - 1);
    int b = plane >> 8;
    int base  = plane * HWSZ;
    int nbase = b * HWSZ;
    float acc = 0.0f;
    for (int it = 0; it < 16; ++it) {
        int hw = it * 256 + threadIdx.x;
        int id = idx[nbase + hw];
        float v = ze[base + hw];
        float q = cb[id * DDIM + d];
        zq[base + hw] = q;                  // straight-through forward value == z_q
        float df = v - q;
        acc += df * df;
        atomicAdd(&dw[id * DDIM + d], v);
    }
    __shared__ float red[256];
    red[threadIdx.x] = acc;
    __syncthreads();
    for (int s = 128; s > 0; s >>= 1) {
        if (threadIdx.x < s) red[threadIdx.x] += red[threadIdx.x + s];
        __syncthreads();
    }
    if (threadIdx.x == 0) atomicAdd(lossAcc, red[0]);
}

// ---------------- kernel 5: EMA finalize ----------------
__global__ __launch_bounds__(256) void k_final(const float* __restrict__ emaCnt,
                                               const float* __restrict__ emaW,
                                               const float* __restrict__ counts,
                                               const float* __restrict__ dw,
                                               const float* __restrict__ lossAcc,
                                               float* __restrict__ outCb,
                                               float* __restrict__ outCnt,
                                               float* __restrict__ outW,
                                               float* __restrict__ outLoss) {
    int k = blockIdx.x, d = threadIdx.x;
    int kd = k * DDIM + d;
    float nc = emaCnt[k] * DECAYF + (1.0f - DECAYF) * counts[k];
    float nw = emaW[kd] * DECAYF + (1.0f - DECAYF) * dw[kd];
    outW[kd]  = nw;
    outCb[kd] = nw / (nc + EPSF);
    if (d == 0) outCnt[k] = nc;
    if (k == 0 && d == 0) outLoss[0] = lossAcc[0] * (1.0f / 16777216.0f); // mean over B*D*H*W
}

extern "C" void kernel_launch(void* const* d_in, const int* in_sizes, int n_in,
                              void* d_out, int out_size, void* d_ws, size_t ws_size,
                              hipStream_t stream) {
    (void)in_sizes; (void)n_in; (void)out_size; (void)ws_size;
    const float* ze     = (const float*)d_in[0];   // [16,256,64,64]
    const float* cb     = (const float*)d_in[1];   // [2048,256]
    const float* emaCnt = (const float*)d_in[2];   // [2048]
    const float* emaW   = (const float*)d_in[3];   // [2048,256]

    // workspace layout (bytes)
    char* ws = (char*)d_ws;
    unsigned short* cbb = (unsigned short*)(ws);                       // 1,048,576 B
    float* cnorm   = (float*)(ws + 1048576);                           //     8,192 B
    float* counts  = (float*)(ws + 1056768);                           //     8,192 B
    float* dw      = (float*)(ws + 1064960);                           // 2,097,152 B
    float* lossAcc = (float*)(ws + 3162112);                           //         4 B
    int*   wsIdx   = (int*)  (ws + 3162116);                           //   262,144 B

    // output layout (floats, tuple flattened in return order)
    float* out     = (float*)d_out;
    float* outZq   = out;                  // 4,194,304
    float* outIdx  = out + 4194304;        //    65,536
    float* outLoss = out + 4259840;        //         1
    float* outCb   = out + 4259841;        //   524,288
    float* outCnt  = out + 4784129;        //     2,048
    float* outW    = out + 4786177;        //   524,288

    // counts + dw + loss are contiguous: zero in one shot
    int nzero = KCODES + KCODES * DDIM + 1;                            // 526,337
    k_zero<<<(nzero + 255) / 256, 256, 0, stream>>>(counts, nzero);

    k_pre<<<KCODES, 256, 0, stream>>>(cb, cbb, cnorm);

    k_argmin<<<NVEC / ROWS, 128, 0, stream>>>(ze, cbb, cnorm, wsIdx, outIdx, counts);

    k_zq<<<BATCH * DDIM, 256, 0, stream>>>(ze, cb, wsIdx, outZq, dw, lossAcc);

    k_final<<<KCODES, 256, 0, stream>>>(emaCnt, emaW, counts, dw, lossAcc,
                                        outCb, outCnt, outW, outLoss);
}
